// CNLB_46145128628249
// MI455X (gfx1250) — matrix operand order repaired and checked by reading, not compiled, not run on hardware
//
#include <hip/hip_runtime.h>

// ---- CDNA5 (gfx1250) wave32 WMMA types ----
typedef __bf16 bf16;
typedef __attribute__((ext_vector_type(16))) __bf16 v16bf;
typedef __attribute__((ext_vector_type(8)))  float  v8f;
typedef __attribute__((ext_vector_type(4)))  unsigned int u32x4;
typedef __attribute__((ext_vector_type(4)))  int int32x4;
typedef __attribute__((ext_vector_type(8)))  int int32x8;

union V16U { v16bf v; u32x4 q[2]; };
union P8   { u32x4 q; bf16 h[8]; };

#define CN 64
#define NN 65536
#define NT 128
#define BB 4

#if __has_builtin(__builtin_amdgcn_tensor_load_to_lds) && __has_builtin(__builtin_amdgcn_s_wait_tensorcnt)
#define HAS_TDM 1
#else
#define HAS_TDM 0
#endif

__device__ __forceinline__ v8f wmma_bf16(v16bf a, v16bf b, v8f c) {
  return __builtin_amdgcn_wmma_f32_16x16x32_bf16(false, a, false, b, (short)0, c, false, false);
}

// A-matrix (16-bit 16x32): lanes 0-15 hold K k0..k0+15; lanes 16-31 +16.
__device__ __forceinline__ v16bf load_A(const bf16* W, int m0, int k0, int lh, int hi, int rs) {
  const bf16* row = W + (m0 + lh) * rs;
  const int d = hi * 16;
  V16U u;
  u.q[0] = *(const u32x4*)(row + k0 + d);
  u.q[1] = *(const u32x4*)(row + k0 + 8 + d);
  return u.v;
}

// B-matrix (16-bit 32x16) from a [n][k] row: N=lane&15; lanes 0-15 K k0..k0+15, 16-31 +16.
__device__ __forceinline__ v16bf load_B(const bf16* colrow, int k0, int hi) {
  const bf16* p = colrow + k0 + hi * 16;
  V16U u;
  u.q[0] = *(const u32x4*)(p);
  u.q[1] = *(const u32x4*)(p + 8);
  return u.v;
}

// B-matrix built from a column of a row-major [k][NT] f32 LDS tile (TDM raw layout).
__device__ __forceinline__ v16bf load_B_colF32(const float* X, int col, int kbase) {
  v16bf r;
  #pragma unroll
  for (int kk = 0; kk < 16; ++kk) r[kk] = (bf16)X[(kbase + kk) * NT + col];
  return r;
}
__device__ __forceinline__ v16bf load_B_colBF(const bf16* X, int col, int kbase) {
  v16bf r;
  #pragma unroll
  for (int kk = 0; kk < 16; ++kk) r[kk] = X[(kbase + kk) * NT + col];
  return r;
}

#if HAS_TDM
// One-instruction DMA of a 2D tile (tile_d0 x tile_d1 elems, row stride stride0 elems)
// from global to LDS.  D# packed per ISA 8.3/8.4 (group0: count=1,type=2; group1: dims).
// amdgpu-toolchain (clang-23) 6-arg form: (g0, g1, g2, g3, extra int32x8, cpol).
__device__ __forceinline__ void tdm_load_2d(unsigned long long ga, unsigned int lds,
                                            int dsz_code, int tile_d0, int tile_d1,
                                            unsigned int tensor_d0, unsigned int tensor_d1,
                                            unsigned int stride0) {
  u32x4 g0;
  g0[0] = 1u;                                                    // count=1, user D#
  g0[1] = lds;                                                   // LDS byte address
  g0[2] = (unsigned int)(ga & 0xFFFFFFFFu);                      // global_addr[31:0]
  g0[3] = (unsigned int)((ga >> 32) & 0x01FFFFFFu) | (2u << 30); // addr[56:32] | type=2
  int32x8 g1;
  g1[0] = dsz_code << 16;                                        // data_size, no flags
  g1[1] = (int)((tensor_d0 & 0xFFFFu) << 16);                    // tensor_dim0[15:0]
  g1[2] = (int)(((tensor_d0 >> 16) & 0xFFFFu) | ((tensor_d1 & 0xFFFFu) << 16));
  g1[3] = (int)((((tensor_d1 >> 16) & 0xFFFFu)) | ((unsigned)tile_d0 << 16));
  g1[4] = tile_d1 & 0xFFFF;                                      // tile_dim1 | tile_dim2=0
  g1[5] = (int)stride0;                                          // dim0_stride[31:0]
  g1[6] = 0;
  g1[7] = 0;
  int32x4 z4 = {0, 0, 0, 0};
  int32x8 z8 = {0, 0, 0, 0, 0, 0, 0, 0};
  __builtin_amdgcn_tensor_load_to_lds(g0, g1, z4, z4, z8, 0);
}
#endif

// ---------- K0: zero accumulators + convert weights to bf16 ----------
__global__ void __launch_bounds__(256) k_init(
    const float* __restrict__ thw, const float* __restrict__ phw,
    const float* __restrict__ gw,  const float* __restrict__ ww,
    bf16* __restrict__ wbf, float* __restrict__ fbuf, float* __restrict__ stats) {
  int idx = blockIdx.x * 256 + threadIdx.x;
  if (idx < BB * 64 * 64) fbuf[idx] = 0.0f;
  if (idx < 128) stats[idx] = 0.0f;
  if (idx < 4096) {
    wbf[idx]         = (bf16)thw[idx];
    wbf[4096 + idx]  = (bf16)phw[idx];
    wbf[8192 + idx]  = (bf16)gw[idx];
    wbf[12288 + idx] = (bf16)ww[idx];
  }
}

// ---------- K1: fused theta/phi/g convs + partial f = phi * theta^T ----------
__global__ void __launch_bounds__(256) k_qkv_f(
    const float* __restrict__ x, const bf16* __restrict__ wbf,
    const float* __restrict__ bth, const float* __restrict__ bph, const float* __restrict__ bg,
    bf16* __restrict__ gout /* [b][n][c] bf16 */, float* __restrict__ fbuf) {
  __shared__ __align__(16) float xF[CN * NT];   // raw x tile [c][j] f32 (TDM dest)
  __shared__ __align__(16) bf16 thT[CN * NT];   // theta tile [c][n]
  __shared__ __align__(16) bf16 phT[CN * NT];   // phi tile   [c][n]

  const int b = blockIdx.y, tid = threadIdx.x;
  const int wid = tid >> 5, lane = tid & 31, lh = lane & 15, hi = lane >> 4;
  const int nw = wid * 16;

  const float* xb = x + (size_t)b * CN * NN;
  bf16* gb = gout + (size_t)b * CN * NN;
  const bf16* Wth = wbf;
  const bf16* Wph = wbf + 4096;
  const bf16* Wg  = wbf + 8192;

  const int t0 = wid * 2;
  const int ct0 = t0 >> 2,       dt0 = t0 & 3;
  const int ct1 = (t0 + 1) >> 2, dt1 = (t0 + 1) & 3;
  v8f fc0 = {0,0,0,0,0,0,0,0}, fc1 = {0,0,0,0,0,0,0,0};

  for (int it = 0; it < 8; ++it) {
    const int n0 = (blockIdx.x * 8 + it) * NT;
    __syncthreads();
#if HAS_TDM
    if (tid < 32) {   // wave 0 issues one DMA for the whole [64 x 128] f32 tile
      tdm_load_2d((unsigned long long)(uintptr_t)(xb + n0),
                  (unsigned int)(uintptr_t)xF,
                  /*dsz=4B*/2, NT, CN, NN, CN, NN);
      __builtin_amdgcn_s_wait_tensorcnt(0);
    }
#else
    for (int i = tid; i < CN * NT; i += 256) {
      int c = i >> 7, j = i & (NT - 1);
      xF[c * NT + j] = xb[(size_t)c * NN + n0 + j];
    }
#endif
    __syncthreads();

    const int col = nw + lh;
    v16bf xv0 = load_B_colF32(xF, col, 0 + hi * 16);
    v16bf xv1 = load_B_colF32(xF, col, 32 + hi * 16);

    // g conv -> global bf16 [n][c], packed 16B stores
    #pragma unroll
    for (int mt = 0; mt < 4; ++mt) {
      v8f acc = {0,0,0,0,0,0,0,0};
      acc = wmma_bf16(load_A(Wg, mt*16, 0,  lh, hi, CN), xv0, acc);
      acc = wmma_bf16(load_A(Wg, mt*16, 32, lh, hi, CN), xv1, acc);
      const int mb = mt*16 + hi*8;
      P8 p;
      #pragma unroll
      for (int v = 0; v < 8; ++v) p.h[v] = (bf16)(acc[v] + bg[mb + v]);
      *(u32x4*)(gb + (size_t)(n0 + col) * CN + mb) = p.q;
    }
    // theta -> LDS [c][n]
    #pragma unroll
    for (int mt = 0; mt < 4; ++mt) {
      v8f acc = {0,0,0,0,0,0,0,0};
      acc = wmma_bf16(load_A(Wth, mt*16, 0,  lh, hi, CN), xv0, acc);
      acc = wmma_bf16(load_A(Wth, mt*16, 32, lh, hi, CN), xv1, acc);
      const int mb = mt*16 + hi*8;
      #pragma unroll
      for (int v = 0; v < 8; ++v)
        thT[(mb + v) * NT + nw + lh] = (bf16)(acc[v] + bth[mb + v]);
    }
    // phi -> LDS [c][n]
    #pragma unroll
    for (int mt = 0; mt < 4; ++mt) {
      v8f acc = {0,0,0,0,0,0,0,0};
      acc = wmma_bf16(load_A(Wph, mt*16, 0,  lh, hi, CN), xv0, acc);
      acc = wmma_bf16(load_A(Wph, mt*16, 32, lh, hi, CN), xv1, acc);
      const int mb = mt*16 + hi*8;
      #pragma unroll
      for (int v = 0; v < 8; ++v)
        phT[(mb + v) * NT + nw + lh] = (bf16)(acc[v] + bph[mb + v]);
    }
    __syncthreads();

    // partial f += phi(ct,:) * theta(dt,:)^T over K = NT columns
    const bf16* b0row = &thT[(dt0*16 + lh) * NT];
    const bf16* b1row = &thT[(dt1*16 + lh) * NT];
    #pragma unroll
    for (int k0 = 0; k0 < NT; k0 += 32) {
      fc0 = wmma_bf16(load_A(phT, ct0*16, k0, lh, hi, NT), load_B(b0row, k0, hi), fc0);
      fc1 = wmma_bf16(load_A(phT, ct1*16, k0, lh, hi, NT), load_B(b1row, k0, hi), fc1);
    }
  }

  float* fb = fbuf + b * 4096;
  #pragma unroll
  for (int v = 0; v < 8; ++v) {
    atomicAdd(&fb[(ct0*16 + hi*8 + v) * 64 + dt0*16 + lh], fc0[v]);
    atomicAdd(&fb[(ct1*16 + hi*8 + v) * 64 + dt1*16 + lh], fc1[v]);
  }
}

// ---------- K2: row softmax over f[b,c,:] -> bf16 attn ----------
__global__ void __launch_bounds__(256) k_softmax(const float* __restrict__ fbuf,
                                                 bf16* __restrict__ attnbf) {
  int r = threadIdx.x;
  if (r >= BB * 64) return;
  const float* row = fbuf + r * 64;
  float m = -3.4e38f;
  for (int d = 0; d < 64; ++d) m = fmaxf(m, row[d]);
  float s = 0.0f;
  for (int d = 0; d < 64; ++d) s += __expf(row[d] - m);
  const float inv = 1.0f / s;
  bf16* arow = attnbf + r * 64;
  for (int d = 0; d < 64; ++d) arow[d] = (bf16)(__expf(row[d] - m) * inv);
}

// ---------- K3: y = attn @ g; permute folded into [n][c] bf16 stores ----------
__global__ void __launch_bounds__(256) k_y(const bf16* __restrict__ gin /* [b][n][c] */,
                                           const bf16* __restrict__ attnbf,
                                           bf16* __restrict__ yperm /* [b][n][c] */) {
  __shared__ __align__(16) bf16 gT[NT * CN];    // [j][c] — contiguous 16KB slice of gin
  const int b = blockIdx.y, n0 = blockIdx.x * NT, tid = threadIdx.x;
  const bf16* gb = gin + (size_t)b * CN * NN;
#if HAS_TDM
  if (tid < 32) {   // contiguous 8192-element 1D tile
    tdm_load_2d((unsigned long long)(uintptr_t)(gb + (size_t)n0 * CN),
                (unsigned int)(uintptr_t)gT,
                /*dsz=2B*/1, CN * NT, 0, CN * NT, 1, CN * NT);
    __builtin_amdgcn_s_wait_tensorcnt(0);
  }
#else
  for (int i = tid; i < CN * NT; i += 256) gT[i] = gb[(size_t)n0 * CN + i];
#endif
  __syncthreads();
  const int wid = tid >> 5, lane = tid & 31, lh = lane & 15, hi = lane >> 4;
  const int nw = wid * 16;
  const bf16* A = attnbf + b * 4096;
  const bf16* brow = &gT[(nw + lh) * CN];       // K = d runs along the row
  v16bf gv0 = load_B(brow, 0, hi);
  v16bf gv1 = load_B(brow, 32, hi);
  bf16* yb = yperm + (size_t)b * CN * NN;
  const int n = n0 + nw + lh;
  #pragma unroll
  for (int mt = 0; mt < 4; ++mt) {
    v8f acc = {0,0,0,0,0,0,0,0};
    acc = wmma_bf16(load_A(A, mt*16, 0,  lh, hi, CN), gv0, acc);
    acc = wmma_bf16(load_A(A, mt*16, 32, lh, hi, CN), gv1, acc);
    const int mb = mt*16 + hi*8;
    P8 p;
    #pragma unroll
    for (int v = 0; v < 8; ++v) p.h[v] = (bf16)acc[v];
    *(u32x4*)(yb + (size_t)n * CN + mb) = p.q;   // flat m = n*64 + c (reshape quirk)
  }
}

// ---------- K4: w conv on permuted y + BN sum/sumsq ----------
__global__ void __launch_bounds__(256) k_wconv(const bf16* __restrict__ yperm,
                                               const bf16* __restrict__ Ww,
                                               const float* __restrict__ wb,
                                               float* __restrict__ wy,
                                               float* __restrict__ stats) {
  __shared__ __align__(16) bf16 yT[CN * NT];    // [co][j] raw strided tile (TDM dest)
  __shared__ float ssum[CN], ssq[CN];
  const int b = blockIdx.y, n0 = blockIdx.x * NT, tid = threadIdx.x;
  if (tid < CN) { ssum[tid] = 0.0f; ssq[tid] = 0.0f; }
  const bf16* yb = yperm + (size_t)b * CN * NN;  // viewed as [co][p], row stride NN
#if HAS_TDM
  if (tid < 32) {
    tdm_load_2d((unsigned long long)(uintptr_t)(yb + n0),
                (unsigned int)(uintptr_t)yT,
                /*dsz=2B*/1, NT, CN, NN, CN, NN);
    __builtin_amdgcn_s_wait_tensorcnt(0);
  }
#else
  for (int i = tid; i < CN * NT; i += 256) {
    int c = i >> 7, j = i & (NT - 1);
    yT[c * NT + j] = yb[(size_t)c * NN + n0 + j];
  }
#endif
  __syncthreads();
  const int wid = tid >> 5, lane = tid & 31, lh = lane & 15, hi = lane >> 4;
  const int nw = wid * 16;
  const int col = nw + lh;
  v16bf yv0 = load_B_colBF(yT, col, 0 + hi * 16);
  v16bf yv1 = load_B_colBF(yT, col, 32 + hi * 16);
  float* ob = wy + (size_t)b * CN * NN + n0 + col;
  #pragma unroll
  for (int mt = 0; mt < 4; ++mt) {
    v8f acc = {0,0,0,0,0,0,0,0};
    acc = wmma_bf16(load_A(Ww, mt*16, 0,  lh, hi, CN), yv0, acc);
    acc = wmma_bf16(load_A(Ww, mt*16, 32, lh, hi, CN), yv1, acc);
    const int mb = mt*16 + hi*8;
    #pragma unroll
    for (int v = 0; v < 8; ++v) {
      const int m = mb + v;
      const float val = acc[v] + wb[m];
      ob[(size_t)m * NN] = val;
      atomicAdd(&ssum[m], val);
      atomicAdd(&ssq[m], val * val);
    }
  }
  __syncthreads();
  if (tid < CN) {
    atomicAdd(&stats[tid], ssum[tid]);
    atomicAdd(&stats[CN + tid], ssq[tid]);
  }
}

// ---------- K5: finalize BN scale/shift ----------
__global__ void k_bnfin(const float* __restrict__ stats, const float* __restrict__ gamma,
                        const float* __restrict__ beta, float* __restrict__ ss) {
  int c = threadIdx.x;
  if (c >= CN) return;
  const float cnt = (float)BB * (float)NN;
  const float mean = stats[c] / cnt;
  const float var  = stats[CN + c] / cnt - mean * mean;
  const float sc   = gamma[c] * rsqrtf(var + 1e-5f);
  ss[c] = sc;
  ss[CN + c] = beta[c] - mean * sc;
}

// ---------- K6: out = wy*scale + shift + x (float4) ----------
__global__ void __launch_bounds__(256) k_final(const float* __restrict__ wy,
                                               const float* __restrict__ x,
                                               const float* __restrict__ ss,
                                               float* __restrict__ out) {
  const size_t total = (size_t)BB * CN * NN / 4;
  for (size_t i = (size_t)blockIdx.x * 256 + threadIdx.x; i < total;
       i += (size_t)gridDim.x * 256) {
    const size_t e = i * 4;
    const int c = (int)((e >> 16) & 63);
    const float4 w4 = ((const float4*)wy)[i];
    const float4 x4 = ((const float4*)x)[i];
    const float sc = ss[c], sh = ss[CN + c];
    float4 o;
    o.x = w4.x * sc + sh + x4.x;
    o.y = w4.y * sc + sh + x4.y;
    o.z = w4.z * sc + sh + x4.z;
    o.w = w4.w * sc + sh + x4.w;
    ((float4*)out)[i] = o;
  }
}

extern "C" void kernel_launch(void* const* d_in, const int* in_sizes, int n_in,
                              void* d_out, int out_size, void* d_ws, size_t ws_size,
                              hipStream_t stream) {
  const float* x     = (const float*)d_in[0];
  const float* thw   = (const float*)d_in[1];
  const float* thb   = (const float*)d_in[2];
  const float* phw   = (const float*)d_in[3];
  const float* phb   = (const float*)d_in[4];
  const float* gw    = (const float*)d_in[5];
  const float* gbi   = (const float*)d_in[6];
  const float* ww    = (const float*)d_in[7];
  const float* wbi   = (const float*)d_in[8];
  const float* gamma = (const float*)d_in[9];
  const float* beta  = (const float*)d_in[10];

  const size_t BCN = (size_t)BB * CN * NN;
  float* wy     = (float*)d_ws;                       // [B,64,65536] f32
  float* fbuf   = wy + BCN;                           // [B,64,64]
  float* stats  = fbuf + (size_t)BB * 64 * 64;        // sum|sumsq
  float* ss     = stats + 128;                        // scale|shift
  bf16*  wbf    = (bf16*)(ss + 128);                  // theta|phi|g|w weights
  bf16*  attnbf = wbf + 4 * 4096;                     // [B,64,64]
  bf16*  g      = attnbf + (size_t)BB * 4096;         // [B,N,64] bf16 (transposed)
  bf16*  yp     = g + BCN;                            // [B,N,64] bf16 (= NCHW w_y input)
  float* out    = (float*)d_out;

  k_init   <<<64, 256, 0, stream>>>(thw, phw, gw, ww, wbf, fbuf, stats);
  k_qkv_f  <<<dim3(64, BB), 256, 0, stream>>>(x, wbf, thb, phb, gbi, g, fbuf);
  k_softmax<<<1, 256, 0, stream>>>(fbuf, attnbf);
  k_y      <<<dim3(NN / NT, BB), 256, 0, stream>>>(g, attnbf, yp);
  k_wconv  <<<dim3(NN / NT, BB), 256, 0, stream>>>(yp, wbf + 3 * 4096, wbi, wy, stats);
  k_bnfin  <<<1, 64, 0, stream>>>(stats, gamma, beta, ss);
  k_final  <<<4096, 256, 0, stream>>>(wy, x, ss, out);
}
